// TensorSquare_36636071035615
// MI455X (gfx1250) — compile-verified, matrix operand order checked by reading, not run on hardware
//
#include <hip/hip_runtime.h>
#include <hip/hip_bf16.h>

// ---------------------------------------------------------------------------
// TensorSquare: out[z,k] = sum_{i,j} C[k, i*128+j] * f[z,i] * f[z,j]
// bf16 WMMA GEMM with A (outer products) generated in-register from an
// LDS-resident feature tile. B chunks are DMA'd into LDS by the CDNA5
// Tensor Data Mover (double-buffered, TENSORcnt-tracked). Inputs are
// pre-converted f32->bf16 into d_ws by a bandwidth-bound prep kernel.
// Fallback (small d_ws): round-1 VGPR-staged kernel.
// ---------------------------------------------------------------------------

typedef __attribute__((ext_vector_type(16))) __bf16   v16bf;
typedef __attribute__((ext_vector_type(8)))  float    v8f;
typedef __attribute__((ext_vector_type(8)))  unsigned v8u;
typedef __attribute__((ext_vector_type(4)))  unsigned v4u;
typedef __attribute__((ext_vector_type(8)))  int      v8i;
typedef __attribute__((ext_vector_type(4)))  int      v4i;

#define ZDIM 16384
#define NDIM 128
#define KDIM 1024

// LDS layout (bytes). Rows padded by 16B to spread fragment loads over banks;
// the TDM applies the same padding in hardware (pad_interval/pad_amount).
#define FSH_STRIDE 272                 // 128 bf16 (256B) + 16B pad
#define BSH_STRIDE 80                  // 32 bf16 (64B)  + 16B pad
#define FSH_BYTES (128 * FSH_STRIDE)   // 34816
#define BSH_BYTES (128 * BSH_STRIDE)   // 10240

static __device__ __forceinline__ unsigned pack_bf16(float a, float b) {
    __bf16 lo = (__bf16)a;
    __bf16 hi = (__bf16)b;
    unsigned short ul = __builtin_bit_cast(unsigned short, lo);
    unsigned short uh = __builtin_bit_cast(unsigned short, hi);
    return ((unsigned)uh << 16) | (unsigned)ul;
}

static __device__ __forceinline__ unsigned pk_mul_bf16(unsigned a, unsigned b) {
    unsigned d;
    asm("v_pk_mul_bf16 %0, %1, %2" : "=v"(d) : "v"(a), "v"(b));
    return d;
}

// ---- TDM helpers ----------------------------------------------------------
// D# group 0: count=1 (valid), lds_addr[63:32], global_addr[120:64],
// type=2 ("image") at bits [127:126].
static __device__ __forceinline__ v4u tdm_g0(unsigned lds_addr, const void* g) {
    unsigned long long ga = (unsigned long long)(uintptr_t)g;
    v4u d;
    d[0] = 1u;
    d[1] = lds_addr;
    d[2] = (unsigned)ga;
    d[3] = (unsigned)((ga >> 32) & 0x01ffffffu) | 0x80000000u;
    return d;
}

static __device__ __forceinline__ void tdm_load(v4u g0, v8i g1) {
    v4i z4 = {0, 0, 0, 0};
#if defined(__has_include)
#if __has_include(<hip/amd_detail/amd_gfx1250_TDM.h>)
    v8i z8 = {0, 0, 0, 0, 0, 0, 0, 0};
    __builtin_amdgcn_tensor_load_to_lds(g0, g1, z4, z4, z8, 0);
#else
    __builtin_amdgcn_tensor_load_to_lds(g0, g1, z4, z4, 0);
#endif
#else
    __builtin_amdgcn_tensor_load_to_lds(g0, g1, z4, z4, 0);
#endif
}

// ---------------------------------------------------------------------------
// Prep: f32 -> bf16, 8 elements/thread (memory-bound)
// ---------------------------------------------------------------------------
__global__ __launch_bounds__(256)
void cvt_f32_to_bf16_x8(const float* __restrict__ src,
                        unsigned* __restrict__ dst, int n8) {
    int i = blockIdx.x * 256 + threadIdx.x;
    if (i >= n8) return;
    float4 a = ((const float4*)src)[2 * i + 0];
    float4 b = ((const float4*)src)[2 * i + 1];
    uint4 o;
    o.x = pack_bf16(a.x, a.y);
    o.y = pack_bf16(a.z, a.w);
    o.z = pack_bf16(b.x, b.y);
    o.w = pack_bf16(b.z, b.w);
    ((uint4*)dst)[i] = o;
}

// ---------------------------------------------------------------------------
// Main GEMM: TDM-staged, double-buffered B, bf16 inputs from d_ws
// ---------------------------------------------------------------------------
__global__ __launch_bounds__(256)
void tensor_square_wmma_tdm(const unsigned short* __restrict__ featbf, // [Z,128]
                            const unsigned short* __restrict__ mixbf,  // [K,16384]
                            float* __restrict__ out) {
    __shared__ __align__(16) unsigned char lds[FSH_BYTES + 2 * BSH_BYTES];
    unsigned char* fsh = lds;
    unsigned char* bsh = lds + FSH_BYTES;

    const int tid  = threadIdx.x;
    const int lane = tid & 31;
    const int wave = tid >> 5;
    const int wm   = wave & 3;    // M group: rows 32*wm .. +31
    const int wn   = wave >> 2;   // N group: cols 64*wn .. +63
    const int h    = lane >> 4;   // lane half (K sub-block select, ISA layout)
    const int ml   = lane & 15;   // M (A) / N (B) index within 16x16 subtile

    const int z0 = blockIdx.x * 128;
    const int k0 = blockIdx.y * 128;

    const unsigned ldsbase = (unsigned)(uintptr_t)&lds[0];

    // D# group 1 descriptors (uniform).
    // feat tile: 128x128 bf16, row 256B (64 DW) + 16B pad -> stride 272B
    //   d0: data_size=1(2B) | pad_en | pad_interval=5(64DW) | pad_amount=3(4DW)
    const v8i g1_feat = {0x07510000, (int)(128u << 16), (int)(16384u << 16),
                         (int)(128u << 16), 128, 128, 0, 0};
    // B chunk: 128x32 bf16, row 64B (16 DW) + 16B pad -> stride 80B
    //   d0: data_size=1 | pad_en | pad_interval=3(16DW) | pad_amount=3(4DW)
    const v8i g1_b = {0x06D10000, (int)(16384u << 16), (int)(1024u << 16),
                      (int)(32u << 16), 128, 16384, 0, 0};

    const unsigned short* mixblk = mixbf + (size_t)k0 * (NDIM * NDIM);

    // ---- Prologue: DMA feature tile + B chunk for step 0 ----
    if (wave == 0) {
        tdm_load(tdm_g0(ldsbase, featbf + (size_t)z0 * NDIM), g1_feat);
        tdm_load(tdm_g0(ldsbase + FSH_BYTES, mixblk), g1_b);
        __builtin_amdgcn_s_wait_tensorcnt(0);
    }
    __syncthreads();

    v8f acc[2][4];
#pragma unroll
    for (int t = 0; t < 2; ++t)
#pragma unroll
        for (int c = 0; c < 4; ++c)
            acc[t][c] = (v8f){0.f, 0.f, 0.f, 0.f, 0.f, 0.f, 0.f, 0.f};

    v8u araw[2];

    // 512 K-steps of 32: step s -> jc = s>>7, i = s&127, p0 = i*128 + jc*32
    for (int s = 0; s < 512; ++s) {
        const int jc = s >> 7;
        const int i  = s & 127;

        // Raw A j-vectors depend only on jc: (re)load 4x total.
        // A fragment layout (16-bit A 16x32): lane half h holds K=8h..8h+7
        // (bytes 16h..) and K=16+8h..+7 (bytes 32+16h..).
        if (i == 0) {
#pragma unroll
            for (int t = 0; t < 2; ++t) {
                const int zl = 32 * wm + 16 * t + ml;
                const unsigned char* p =
                    fsh + zl * FSH_STRIDE + jc * 64 + 16 * h;
                uint4 lo = *(const uint4*)(p);
                uint4 hi = *(const uint4*)(p + 32);
                araw[t] = (v8u){lo.x, lo.y, lo.z, lo.w, hi.x, hi.y, hi.z, hi.w};
            }
        }

        // ---- Kick off DMA for step s+1 into the other buffer (overlapped) ----
        if (wave == 0 && s + 1 < 512) {
            const int s1 = s + 1;
            const int p1 = (s1 & 127) * NDIM + (s1 >> 7) * 32;
            tdm_load(tdm_g0(ldsbase + FSH_BYTES + (unsigned)((s1 & 1) * BSH_BYTES),
                            mixblk + p1),
                     g1_b);
        }

        const unsigned char* bb = bsh + (s & 1) * BSH_BYTES;

        // ---- A fragments: scale raw j-vector by splat bf16 f[z, i] ----
        v16bf a[2];
#pragma unroll
        for (int t = 0; t < 2; ++t) {
            const int zl = 32 * wm + 16 * t + ml;
            unsigned short sc =
                *(const unsigned short*)(fsh + zl * FSH_STRIDE + i * 2);
            unsigned s2 = (unsigned)sc * 0x00010001u;  // [s, s] packed
            v8u av;
#pragma unroll
            for (int q = 0; q < 8; ++q) av[q] = pk_mul_bf16(araw[t][q], s2);
            a[t] = __builtin_bit_cast(v16bf, av);
        }

        // ---- B fragments: lane ml is the N column; same K sub-block split ----
        v16bf b[4];
#pragma unroll
        for (int c = 0; c < 4; ++c) {
            const int kl = 64 * wn + 16 * c + ml;
            const unsigned char* p = bb + kl * BSH_STRIDE + 16 * h;
            uint4 lo = *(const uint4*)(p);
            uint4 hi = *(const uint4*)(p + 32);
            v8u bv = (v8u){lo.x, lo.y, lo.z, lo.w, hi.x, hi.y, hi.z, hi.w};
            b[c] = __builtin_bit_cast(v16bf, bv);
        }

        // ---- 8 WMMAs: D = A x B + C, f32 accumulate ----
#pragma unroll
        for (int t = 0; t < 2; ++t)
#pragma unroll
            for (int c = 0; c < 4; ++c)
                acc[t][c] = __builtin_amdgcn_wmma_f32_16x16x32_bf16(
                    false, a[t], false, b[c], (short)0, acc[t][c],
                    false, false);

        // Next buffer must be fully written before the next step consumes it.
        if (wave == 0) __builtin_amdgcn_s_wait_tensorcnt(0);
        __syncthreads();
    }

    // ---- Store: C/D layout is lanes 0-15 -> M=v, lanes 16-31 -> M=v+8 ----
#pragma unroll
    for (int t = 0; t < 2; ++t) {
#pragma unroll
        for (int c = 0; c < 4; ++c) {
#pragma unroll
            for (int v = 0; v < 8; ++v) {
                const int row = z0 + 32 * wm + 16 * t + v + 8 * h;
                const int col = k0 + 64 * wn + 16 * c + ml;
                out[(size_t)row * KDIM + col] = acc[t][c][v];
            }
        }
    }
}

// ---------------------------------------------------------------------------
// Fallback (round-1): VGPR-staged bf16 conversion, no scratch required
// ---------------------------------------------------------------------------
__global__ __launch_bounds__(256)
void tensor_square_wmma_bf16(const float* __restrict__ feat,
                             const float* __restrict__ mix,
                             float* __restrict__ out) {
    __shared__ __align__(16) unsigned char lds[FSH_BYTES + BSH_BYTES];
    unsigned char* fsh = lds;
    unsigned char* bsh = lds + FSH_BYTES;

    const int tid  = threadIdx.x;
    const int lane = tid & 31;
    const int wave = tid >> 5;
    const int wm   = wave & 3;
    const int wn   = wave >> 2;
    const int h    = lane >> 4;
    const int ml   = lane & 15;

    const int z0 = blockIdx.x * 128;
    const int k0 = blockIdx.y * 128;

    {
        const int r    = tid >> 1;
        const int half = tid & 1;
        const float*   src = feat + (size_t)(z0 + r) * NDIM + half * 64;
        unsigned char* dst = fsh + r * FSH_STRIDE + half * 128;
#pragma unroll
        for (int q = 0; q < 8; ++q) {
            float4 f0 = ((const float4*)src)[2 * q + 0];
            float4 f1 = ((const float4*)src)[2 * q + 1];
            uint4 o;
            o.x = pack_bf16(f0.x, f0.y);
            o.y = pack_bf16(f0.z, f0.w);
            o.z = pack_bf16(f1.x, f1.y);
            o.w = pack_bf16(f1.z, f1.w);
            ((uint4*)dst)[q] = o;
        }
    }
    __syncthreads();

    v8f acc[2][4];
#pragma unroll
    for (int t = 0; t < 2; ++t)
#pragma unroll
        for (int c = 0; c < 4; ++c)
            acc[t][c] = (v8f){0.f, 0.f, 0.f, 0.f, 0.f, 0.f, 0.f, 0.f};

    for (int jc = 0; jc < 4; ++jc) {
        v8u araw[2];
#pragma unroll
        for (int t = 0; t < 2; ++t) {
            const int zl = 32 * wm + 16 * t + ml;
            const unsigned char* p = fsh + zl * FSH_STRIDE + jc * 64 + 16 * h;
            uint4 lo = *(const uint4*)(p);
            uint4 hi = *(const uint4*)(p + 32);
            araw[t] = (v8u){lo.x, lo.y, lo.z, lo.w, hi.x, hi.y, hi.z, hi.w};
        }

        for (int i = 0; i < 128; ++i) {
            __syncthreads();
            {
                const int r    = tid >> 1;
                const int half = tid & 1;
                const float* src = mix + (size_t)(k0 + r) * (NDIM * NDIM)
                                       + i * NDIM + jc * 32 + half * 16;
                unsigned char* dst = bsh + r * BSH_STRIDE + half * 32;
#pragma unroll
                for (int q = 0; q < 2; ++q) {
                    float4 f0 = ((const float4*)src)[2 * q + 0];
                    float4 f1 = ((const float4*)src)[2 * q + 1];
                    uint4 o;
                    o.x = pack_bf16(f0.x, f0.y);
                    o.y = pack_bf16(f0.z, f0.w);
                    o.z = pack_bf16(f1.x, f1.y);
                    o.w = pack_bf16(f1.z, f1.w);
                    ((uint4*)dst)[q] = o;
                }
            }
            __syncthreads();

            v16bf a[2];
#pragma unroll
            for (int t = 0; t < 2; ++t) {
                const int zl = 32 * wm + 16 * t + ml;
                unsigned short sc =
                    *(const unsigned short*)(fsh + zl * FSH_STRIDE + i * 2);
                unsigned s2 = (unsigned)sc * 0x00010001u;
                v8u av;
#pragma unroll
                for (int q = 0; q < 8; ++q) av[q] = pk_mul_bf16(araw[t][q], s2);
                a[t] = __builtin_bit_cast(v16bf, av);
            }

            v16bf b[4];
#pragma unroll
            for (int c = 0; c < 4; ++c) {
                const int kl = 64 * wn + 16 * c + ml;
                const unsigned char* p = bsh + kl * BSH_STRIDE + 16 * h;
                uint4 lo = *(const uint4*)(p);
                uint4 hi = *(const uint4*)(p + 32);
                v8u bv = (v8u){lo.x, lo.y, lo.z, lo.w, hi.x, hi.y, hi.z, hi.w};
                b[c] = __builtin_bit_cast(v16bf, bv);
            }

#pragma unroll
            for (int t = 0; t < 2; ++t)
#pragma unroll
                for (int c = 0; c < 4; ++c)
                    acc[t][c] = __builtin_amdgcn_wmma_f32_16x16x32_bf16(
                        false, a[t], false, b[c], (short)0, acc[t][c],
                        false, false);
        }
    }

#pragma unroll
    for (int t = 0; t < 2; ++t) {
#pragma unroll
        for (int c = 0; c < 4; ++c) {
#pragma unroll
            for (int v = 0; v < 8; ++v) {
                const int row = z0 + 32 * wm + 16 * t + v + 8 * h;
                const int col = k0 + 64 * wn + 16 * c + ml;
                out[(size_t)row * KDIM + col] = acc[t][c][v];
            }
        }
    }
}

extern "C" void kernel_launch(void* const* d_in, const int* in_sizes, int n_in,
                              void* d_out, int out_size, void* d_ws, size_t ws_size,
                              hipStream_t stream) {
    const float* feat = (const float*)d_in[0];   // [16384, 128] f32
    const float* mix  = (const float*)d_in[1];   // [1024, 16384] f32
    float* out        = (float*)d_out;           // [16384, 1024] f32

    const size_t mixbf_bytes  = (size_t)KDIM * NDIM * NDIM * 2;  // 32 MB
    const size_t featbf_bytes = (size_t)ZDIM * NDIM * 2;         //  4 MB

    dim3 grid(ZDIM / 128, KDIM / 128);  // 128 x 8 = 1024 workgroups

    if (ws_size >= mixbf_bytes + featbf_bytes) {
        unsigned short* mixbf  = (unsigned short*)d_ws;
        unsigned short* featbf = (unsigned short*)((char*)d_ws + mixbf_bytes);

        const int n8_mix  = KDIM * NDIM * NDIM / 8;  // 2,097,152
        const int n8_feat = ZDIM * NDIM / 8;         //   262,144
        cvt_f32_to_bf16_x8<<<(n8_mix + 255) / 256, 256, 0, stream>>>(
            mix, (unsigned*)mixbf, n8_mix);
        cvt_f32_to_bf16_x8<<<(n8_feat + 255) / 256, 256, 0, stream>>>(
            feat, (unsigned*)featbf, n8_feat);

        tensor_square_wmma_tdm<<<grid, dim3(256), 0, stream>>>(featbf, mixbf, out);
    } else {
        tensor_square_wmma_bf16<<<grid, dim3(256), 0, stream>>>(feat, mix, out);
    }
}
